// Attention_80436147519951
// MI455X (gfx1250) — matrix operand order repaired and checked by reading, not compiled, not run on hardware
//
#include <hip/hip_runtime.h>

typedef __bf16 bf16_t;
typedef __attribute__((ext_vector_type(16))) __bf16   v16bf;
typedef __attribute__((ext_vector_type(2)))  __bf16   v2bf;
typedef __attribute__((ext_vector_type(8)))  float    v8f;
typedef __attribute__((ext_vector_type(4)))  float    v4f;
typedef __attribute__((ext_vector_type(8)))  unsigned v8u;
typedef __attribute__((ext_vector_type(4)))  unsigned v4u;

#define BH    32      // B*H = 2*16
#define SDIM  2048
#define DDIM  64
#define KBLK  64      // keys per iteration (4 x 16-key tiles)
#define QBLK  128     // 4 waves x 32 query rows (2 N-tiles per wave)

// Cross-half (lane ^ 16) exchange via v_permlanex16_b32 (identity selectors).
static __device__ __forceinline__ int xhalf_i(int x) {
    return __builtin_amdgcn_permlanex16(x, x, 0x76543210, 0xfedcba98, false, false);
}
static __device__ __forceinline__ float xhalf_f(float x) {
    return __builtin_bit_cast(float, xhalf_i(__builtin_bit_cast(int, x)));
}
static __device__ __forceinline__ int pack2(float a, float b) {
    v2bf p; p[0] = (bf16_t)a; p[1] = (bf16_t)b;       // -> v_cvt_pk_bf16_f32
    return __builtin_bit_cast(int, p);
}
// A-fragment (16x32 16-bit): lane (r,hi) holds K = 16*hi + i at element i.
// base points at row + 16*hi; elements [0..15] at base[0..15].
static __device__ __forceinline__ v16bf load_afrag(const bf16_t* base) {
    const v4u lo = *(const v4u*)(base);
    const v4u hh = *(const v4u*)(base + 8);
    v8u t;
    t[0] = lo[0]; t[1] = lo[1]; t[2] = lo[2]; t[3] = lo[3];
    t[4] = hh[0]; t[5] = hh[1]; t[6] = hh[2]; t[7] = hh[3];
    return __builtin_bit_cast(v16bf, t);
}

__global__ __launch_bounds__(128) void flash_attn_bf16_wmma(
    const float* __restrict__ Q, const float* __restrict__ K,
    const float* __restrict__ V, float* __restrict__ O)
{
    __shared__ __align__(32) bf16_t Klds[KBLK][DDIM];  // K rows   (A of S^T = K x Q^T)   8 KB
    __shared__ __align__(32) bf16_t Vlds[DDIM][KBLK];  // V^T rows (A of O^T = V^T x P^T) 8 KB

    const int tid  = threadIdx.x;
    const int w    = tid >> 5;
    const int lane = tid & 31;
    const int r    = lane & 15;
    const int hi   = lane >> 4;

    const int bh    = blockIdx.y;
    const int qbase = blockIdx.x * QBLK;
    const size_t hoff = (size_t)bh * SDIM * DDIM;
    const float* __restrict__ Qh = Q + hoff;
    const float* __restrict__ Kh = K + hoff;
    const float* __restrict__ Vh = V + hoff;
    float* __restrict__ Oh = O + hoff;

    // ---- Q^T resident B-operands for two 16-query N-tiles, pre-scaled by
    // log2(e)/sqrt(D) so softmax runs in the exp2 domain.
    const float qscale = 0.125f * 1.44269504088896340736f;
    v16bf bq[2][2];
    #pragma unroll
    for (int q = 0; q < 2; ++q) {
        const float* qrow = Qh + (size_t)(qbase + w * 32 + q * 16 + r) * DDIM + 16 * hi;
        #pragma unroll
        for (int i = 0; i < 16; ++i) {
            bq[q][0][i] = (bf16_t)(qscale * qrow[i]);
            bq[q][1][i] = (bf16_t)(qscale * qrow[32 + i]);
        }
    }

    v8f o[2][4];                              // O^T accumulators: [qtile][D-tile]
    #pragma unroll
    for (int q = 0; q < 2; ++q)
        #pragma unroll
        for (int d = 0; d < 4; ++d) o[q][d] = (v8f){};
    float mrun[2] = {-3.0e38f, -3.0e38f};     // per-lane per-qtile, log2 domain
    float lrun[2] = {0.0f, 0.0f};

    for (int kb = 0; kb < SDIM; kb += KBLK) {
        __syncthreads();
        // ---- Cooperative staging (128 threads, 64 keys).
        {
            const int krow = tid >> 1, kseg = tid & 1;
            const float* ks = Kh + (size_t)(kb + krow) * DDIM + kseg * 32;
            v16bf k0, k1;
            #pragma unroll
            for (int i = 0; i < 16; ++i) {
                k0[i] = (bf16_t)ks[i];
                k1[i] = (bf16_t)ks[16 + i];
            }
            *(v16bf*)&Klds[krow][kseg * 32 +  0] = k0;
            *(v16bf*)&Klds[krow][kseg * 32 + 16] = k1;

            const int vp = tid & 31;   // key pair (keys 2vp, 2vp+1)
            const int vd = tid >> 5;   // d segment of 16
            const float* va = Vh + (size_t)(kb + 2 * vp) * DDIM + vd * 16;
            const float* vb = va + DDIM;
            #pragma unroll
            for (int i = 0; i < 16; ++i) {
                v2bf p; p[0] = (bf16_t)va[i]; p[1] = (bf16_t)vb[i];
                *(v2bf*)&Vlds[vd * 16 + i][2 * vp] = p;
            }
            if (kb + KBLK < SDIM) {
                __builtin_prefetch(ks + KBLK * DDIM, 0, 3);
                __builtin_prefetch(va + KBLK * DDIM, 0, 3);
            }
        }
        __syncthreads();

        // ---- GEMM1: S^T = K_tile x Q^T. A-fragments shared by both q-tiles.
        v8f s[2][4];
        #pragma unroll
        for (int q = 0; q < 2; ++q)
            #pragma unroll
            for (int t = 0; t < 4; ++t) s[q][t] = (v8f){};
        #pragma unroll
        for (int t = 0; t < 4; ++t) {
            const v16bf a0 = load_afrag(&Klds[16 * t + r][ 0 + 16 * hi]);
            const v16bf a1 = load_afrag(&Klds[16 * t + r][32 + 16 * hi]);
            #pragma unroll
            for (int q = 0; q < 2; ++q) {
                s[q][t] = __builtin_amdgcn_wmma_f32_16x16x32_bf16(false, a0, false, bq[q][0], (short)0, s[q][t], false, false);
                s[q][t] = __builtin_amdgcn_wmma_f32_16x16x32_bf16(false, a1, false, bq[q][1], (short)0, s[q][t], false, false);
            }
        }

        // ---- Online softmax (exp2 domain) + P^T B-operand assembly, per q-tile.
        v16bf bpv[2][2];
        #pragma unroll
        for (int q = 0; q < 2; ++q) {
            float mx = s[q][0][0];
            #pragma unroll
            for (int t = 0; t < 4; ++t)
                #pragma unroll
                for (int v = 0; v < 8; ++v) mx = fmaxf(mx, s[q][t][v]);
            mx = fmaxf(mx, xhalf_f(mx));
            const float mnew  = fmaxf(mrun[q], mx);
            const float alpha = __builtin_amdgcn_exp2f(mrun[q] - mnew);
            mrun[q] = mnew;

            float p[4][8], ls = 0.0f;
            #pragma unroll
            for (int t = 0; t < 4; ++t)
                #pragma unroll
                for (int v = 0; v < 8; ++v) {
                    p[t][v] = __builtin_amdgcn_exp2f(s[q][t][v] - mnew);
                    ls += p[t][v];
                }
            ls += xhalf_f(ls);
            lrun[q] = lrun[q] * alpha + ls;

            #pragma unroll
            for (int d = 0; d < 4; ++d)
                #pragma unroll
                for (int v = 0; v < 8; ++v) o[q][d][v] *= alpha;

            // bpv[q][0] = keys 0..31 (tiles 0/1), bpv[q][1] = keys 32..63 (tiles 2/3).
            v8u bpA, bpB;
            #pragma unroll
            for (int j = 0; j < 4; ++j) {
                const int a0 = pack2(p[0][2 * j], p[0][2 * j + 1]);
                const int a1 = pack2(p[1][2 * j], p[1][2 * j + 1]);
                const int b0 = pack2(p[2][2 * j], p[2][2 * j + 1]);
                const int b1 = pack2(p[3][2 * j], p[3][2 * j + 1]);
                const int xa0 = xhalf_i(a0), xa1 = xhalf_i(a1);
                const int xb0 = xhalf_i(b0), xb1 = xhalf_i(b1);
                bpA[j]     = (unsigned)(hi ? xa1 : a0);
                bpA[4 + j] = (unsigned)(hi ? a1  : xa0);
                bpB[j]     = (unsigned)(hi ? xb1 : b0);
                bpB[4 + j] = (unsigned)(hi ? b1  : xb0);
            }
            bpv[q][0] = __builtin_bit_cast(v16bf, bpA);
            bpv[q][1] = __builtin_bit_cast(v16bf, bpB);
        }

        // ---- GEMM2: O^T += V^T x P^T. A-fragments shared by both q-tiles.
        #pragma unroll
        for (int d = 0; d < 4; ++d) {
            const v16bf av0 = load_afrag(&Vlds[16 * d + r][ 0 + 16 * hi]);
            const v16bf av1 = load_afrag(&Vlds[16 * d + r][32 + 16 * hi]);
            #pragma unroll
            for (int q = 0; q < 2; ++q) {
                o[q][d] = __builtin_amdgcn_wmma_f32_16x16x32_bf16(false, av0, false, bpv[q][0], (short)0, o[q][d], false, false);
                o[q][d] = __builtin_amdgcn_wmma_f32_16x16x32_bf16(false, av1, false, bpv[q][1], (short)0, o[q][d], false, false);
            }
        }
    }

    // ---- Normalize and store: O^T[d][q] -> contiguous 8-float runs along D.
    #pragma unroll
    for (int q = 0; q < 2; ++q) {
        const float inv = 1.0f / lrun[q];
        float* orow = Oh + (size_t)(qbase + w * 32 + q * 16 + r) * DDIM + 8 * hi;
        #pragma unroll
        for (int d = 0; d < 4; ++d) {
            v4f a;
            a[0] = o[q][d][0] * inv; a[1] = o[q][d][1] * inv;
            a[2] = o[q][d][2] * inv; a[3] = o[q][d][3] * inv;
            *(v4f*)(orow + 16 * d) = a;
            a[0] = o[q][d][4] * inv; a[1] = o[q][d][5] * inv;
            a[2] = o[q][d][6] * inv; a[3] = o[q][d][7] * inv;
            *(v4f*)(orow + 16 * d + 4) = a;
        }
    }
}

extern "C" void kernel_launch(void* const* d_in, const int* in_sizes, int n_in,
                              void* d_out, int out_size, void* d_ws, size_t ws_size,
                              hipStream_t stream) {
    const float* Q = (const float*)d_in[0];
    const float* K = (const float*)d_in[1];
    const float* V = (const float*)d_in[2];
    float* O = (float*)d_out;
    dim3 grid(SDIM / QBLK, BH);
    dim3 block(128);
    hipLaunchKernelGGL(flash_attn_bf16_wmma, grid, block, 0, stream, Q, K, V, O);
}
